// CausalSelfAttention_89515708383304
// MI455X (gfx1250) — matrix operand order repaired and checked by reading, not compiled, not run on hardware
//
#include <hip/hip_runtime.h>
#include <stdint.h>

constexpr int D_MODEL = 1024;
constexpr int NUM_HEADS = 16;
constexpr int D_K = 64;
constexpr int SEQ = 2048;
constexpr int BATCH = 2;
constexpr int M_ROWS = BATCH * SEQ;              // 4096
constexpr int OUT_ELEMS = BATCH * SEQ * D_MODEL; // 4194304

typedef __attribute__((ext_vector_type(16))) __bf16 bf16x16;
typedef __attribute__((ext_vector_type(8)))  __bf16 bf16x8;
typedef __attribute__((ext_vector_type(8)))  float  f32x8;

__device__ __forceinline__ __bf16 f2bf(float f) {
  uint32_t u; __builtin_memcpy(&u, &f, 4);
  uint32_t r = u + 0x7FFFu + ((u >> 16) & 1u);   // round-to-nearest-even
  uint16_t h = (uint16_t)(r >> 16);
  __bf16 b; __builtin_memcpy(&b, &h, 2);
  return b;
}

__device__ __forceinline__ bf16x16 combine8(bf16x8 lo, bf16x8 hi) {
  bf16x16 r;
#pragma unroll
  for (int i = 0; i < 8; ++i) { r[i] = lo[i]; r[i + 8] = hi[i]; }
  return r;
}

// A fragment (16x32, row-major source). ISA layout: lanes 0-15 row m=lane,
// K = k0+{0..7} in v0-3 and k0+{16..23} in v4-7; lanes 16-31 row m=lane-16,
// K = k0+{8..15} and k0+{24..31}.
__device__ __forceinline__ bf16x16 load_a16x32(const __bf16* A, int lda, int row,
                                               int k0, int half) {
  const __bf16* p = A + (size_t)row * lda + k0 + half * 8;
  bf16x8 lo = *(const bf16x8*)p;
  bf16x8 hi = *(const bf16x8*)(p + 16);
  return combine8(lo, hi);
}

// B fragment (32x16): source is column-contiguous (element (k,n) at Bm[n*ldb+k]).
// lanes 0-15: col n=lane, K=k0+{0..7},k0+{16..23}; lanes 16-31: col n=lane-16, K=k0+{8..15},k0+{24..31}.
__device__ __forceinline__ bf16x16 load_b32x16(const __bf16* Bm, int ldb, int col,
                                               int k0, int half) {
  const __bf16* p = Bm + (size_t)col * ldb + k0 + half * 8;
  bf16x8 lo = *(const bf16x8*)p;
  bf16x8 hi = *(const bf16x8*)(p + 16);
  return combine8(lo, hi);
}

__device__ __forceinline__ f32x8 wmma_bf16(bf16x16 a, bf16x16 b, f32x8 c) {
  return __builtin_amdgcn_wmma_f32_16x16x32_bf16(false, a, false, b, (short)0, c,
                                                 false, false);
}

// ---------------- conversion / transpose ----------------
__global__ void k_cvt_bf16(const float* __restrict__ src, __bf16* __restrict__ dst, int n) {
  int i = blockIdx.x * blockDim.x + threadIdx.x;
  if (i < n) dst[i] = f2bf(src[i]);
}

// W [in,out] fp32 row-major -> WT [out,in] bf16 row-major
__global__ void k_transpose_bf16(const float* __restrict__ W, __bf16* __restrict__ WT) {
  int i = blockIdx.x * blockDim.x + threadIdx.x;   // i = o*1024 + ii
  int o = i >> 10, ii = i & 1023;
  WT[i] = f2bf(W[(ii << 10) + o]);
}

// ---------------- fused QKV GEMM (two-phase pipelined, no buffer copies) ------
// z=0: Q -> [B,H,S,dk]; z=1: K -> [B,H,S,dk]; z=2: V -> [B,H,dk,S] (transposed)
__global__ void __launch_bounds__(256)
k_qkv_gemm(const __bf16* __restrict__ xb, const __bf16* __restrict__ WT_all,
           const float* __restrict__ bq, const float* __restrict__ bk,
           const float* __restrict__ bv,
           __bf16* __restrict__ Q, __bf16* __restrict__ K, __bf16* __restrict__ VT) {
  const int lane = threadIdx.x & 31, wave = threadIdx.x >> 5;
  const int half = lane >> 4, lm = lane & 15;
  const int z = blockIdx.z;
  const __bf16* WT = WT_all + (size_t)z * D_MODEL * D_MODEL;
  const float* bias = (z == 0) ? bq : (z == 1) ? bk : bv;
  const int r0 = blockIdx.x * 64 + (wave & 3) * 16;
  const int c0 = blockIdx.y * 128 + (wave >> 2) * 64;

  f32x8 acc[4];
#pragma unroll
  for (int t = 0; t < 4; ++t)
#pragma unroll
    for (int v = 0; v < 8; ++v) acc[t][v] = 0.f;

  bf16x16 a0 = load_a16x32(xb, D_MODEL, r0 + lm, 0, half);
  bf16x16 b0[4];
#pragma unroll
  for (int t = 0; t < 4; ++t) b0[t] = load_b32x16(WT, D_MODEL, c0 + t * 16 + lm, 0, half);

  for (int k0 = 0; k0 < D_MODEL; k0 += 64) {
    // phase 0: prefetch k0+32 into buffer 1, compute with buffer 0
    const int k1 = k0 + 32;                              // D_MODEL % 64 == 0 -> valid
    bf16x16 a1 = load_a16x32(xb, D_MODEL, r0 + lm, k1, half);
    bf16x16 b1[4];
#pragma unroll
    for (int t = 0; t < 4; ++t)
      b1[t] = load_b32x16(WT, D_MODEL, c0 + t * 16 + lm, k1, half);
#pragma unroll
    for (int t = 0; t < 4; ++t) acc[t] = wmma_bf16(a0, b0[t], acc[t]);
    // phase 1: prefetch k0+64 (clamped) into buffer 0, compute with buffer 1
    const int k2 = (k0 + 64 < D_MODEL) ? k0 + 64 : k0;
    a0 = load_a16x32(xb, D_MODEL, r0 + lm, k2, half);
#pragma unroll
    for (int t = 0; t < 4; ++t)
      b0[t] = load_b32x16(WT, D_MODEL, c0 + t * 16 + lm, k2, half);
#pragma unroll
    for (int t = 0; t < 4; ++t) acc[t] = wmma_bf16(a1, b1[t], acc[t]);
  }

#pragma unroll
  for (int t = 0; t < 4; ++t) {
    int oc = c0 + t * 16 + lm;
    float bs = bias[oc];
    int h = oc >> 6, d = oc & 63;
#pragma unroll
    for (int v = 0; v < 8; ++v) {
      int gr = r0 + v + half * 8;
      int bb = gr >> 11, s = gr & (SEQ - 1);
      __bf16 val = f2bf(acc[t][v] + bs);
      size_t bh = (size_t)bb * NUM_HEADS + h;
      if (z == 2)      VT[(bh * D_K + d) * SEQ + s] = val;
      else if (z == 0) Q[(bh * SEQ + s) * D_K + d] = val;
      else             K[(bh * SEQ + s) * D_K + d] = val;
    }
  }
}

// ---------------- attention ----------------
// one wave per 16-query tile; scores recomputed via WMMA in each softmax phase
// so attention weights hit HBM exactly once (the roofline-dominant 536 MB write).
__global__ void __launch_bounds__(256)
k_attn(const __bf16* __restrict__ Q, const __bf16* __restrict__ K,
       const __bf16* __restrict__ VT, float* __restrict__ AWbase,
       __bf16* __restrict__ ctxb) {
  __shared__ float lds_e[8][16][33];   // padded: stride 33 avoids bank conflicts
  const int lane = threadIdx.x & 31, wave = threadIdx.x >> 5;
  const int half = lane >> 4, lm = lane & 15;
  const int gw = blockIdx.x * 8 + wave;
  const int bh = gw >> 7;        // 0..31
  const int qt = gw & 127;       // query tile index
  const int q0 = qt * 16;
  const float inv_scale = 0.125f;  // 1/sqrt(64)

  const __bf16* Qp = Q + (size_t)bh * SEQ * D_K;
  const __bf16* Kp = K + (size_t)bh * SEQ * D_K;
  const __bf16* Vp = VT + (size_t)bh * D_K * SEQ;
  float* AW = AWbase + (size_t)bh * SEQ * SEQ + (size_t)q0 * SEQ;

  bf16x16 aq0 = load_a16x32(Qp, D_K, q0 + lm, 0, half);
  bf16x16 aq1 = load_a16x32(Qp, D_K, q0 + lm, 32, half);

  // ---- pass A: row maxima of scaled+masked scores (paired, two-phase) ----
  float rmax[8];
#pragma unroll
  for (int v = 0; v < 8; ++v) rmax[v] = -3.0e38f;
  bf16x16 e0 = load_b32x16(Kp, D_K, lm, 0, half);   // tile 0
  bf16x16 e1 = load_b32x16(Kp, D_K, lm, 32, half);
  for (int j = 0; j <= qt; j += 2) {
    const int ja = (j + 1 <= qt) ? j + 1 : j;
    bf16x16 o0 = load_b32x16(Kp, D_K, ja * 16 + lm, 0, half);
    bf16x16 o1 = load_b32x16(Kp, D_K, ja * 16 + lm, 32, half);
    f32x8 s;
#pragma unroll
    for (int v = 0; v < 8; ++v) s[v] = 0.f;
    s = wmma_bf16(aq0, e0, s);
    s = wmma_bf16(aq1, e1, s);
#pragma unroll
    for (int v = 0; v < 8; ++v) {
      int m = v + half * 8, kc = j * 16 + lm;
      float val = (kc > q0 + m) ? -1e9f : s[v] * inv_scale;
      rmax[v] = fmaxf(rmax[v], val);
    }
    const int jb = (j + 2 <= qt) ? j + 2 : j;
    e0 = load_b32x16(Kp, D_K, jb * 16 + lm, 0, half);
    e1 = load_b32x16(Kp, D_K, jb * 16 + lm, 32, half);
    if (j + 1 <= qt) {                    // wave-uniform branch: EXEC stays full
      f32x8 s2;
#pragma unroll
      for (int v = 0; v < 8; ++v) s2[v] = 0.f;
      s2 = wmma_bf16(aq0, o0, s2);
      s2 = wmma_bf16(aq1, o1, s2);
#pragma unroll
      for (int v = 0; v < 8; ++v) {
        int m = v + half * 8, kc = (j + 1) * 16 + lm;
        float val = (kc > q0 + m) ? -1e9f : s2[v] * inv_scale;
        rmax[v] = fmaxf(rmax[v], val);
      }
    }
  }
#pragma unroll
  for (int off = 1; off < 16; off <<= 1)
#pragma unroll
    for (int v = 0; v < 8; ++v)
      rmax[v] = fmaxf(rmax[v], __shfl_xor(rmax[v], off, 32));

  // ---- pass B: recompute scores -> exp -> row sums + context ----
  float rsum[8];
#pragma unroll
  for (int v = 0; v < 8; ++v) rsum[v] = 0.f;
  f32x8 ctx[4];
#pragma unroll
  for (int t = 0; t < 4; ++t)
#pragma unroll
    for (int v = 0; v < 8; ++v) ctx[t][v] = 0.f;

  for (int jj = 0; jj <= qt; jj += 2) {
#pragma unroll
    for (int sub = 0; sub < 2; ++sub) {
      int j = jj + sub;
      if (j <= qt) {
        f32x8 s;
#pragma unroll
        for (int v = 0; v < 8; ++v) s[v] = 0.f;
        bf16x16 b0 = load_b32x16(Kp, D_K, j * 16 + lm, 0, half);
        bf16x16 b1 = load_b32x16(Kp, D_K, j * 16 + lm, 32, half);
        s = wmma_bf16(aq0, b0, s);
        s = wmma_bf16(aq1, b1, s);
#pragma unroll
        for (int v = 0; v < 8; ++v) {
          int m = v + half * 8, kc = j * 16 + lm;
          float e = (kc <= q0 + m) ? __expf(s[v] * inv_scale - rmax[v]) : 0.f;
          rsum[v] += e;
          lds_e[wave][m][sub * 16 + lm] = e;
        }
      } else {
#pragma unroll
        for (int v = 0; v < 8; ++v)
          lds_e[wave][v + half * 8][sub * 16 + lm] = 0.f;
      }
    }
    __builtin_amdgcn_wave_barrier();   // keep C-layout stores before A-layout reads
    // read P in A-fragment layout (row = lm, K per ISA layout)
    bf16x16 P;
#pragma unroll
    for (int i = 0; i < 8; ++i) {
      P[i]     = f2bf(lds_e[wave][lm][half * 8 + i]);
      P[i + 8] = f2bf(lds_e[wave][lm][16 + half * 8 + i]);
    }
#pragma unroll
    for (int dt = 0; dt < 4; ++dt) {
      bf16x16 bvv = load_b32x16(Vp, SEQ, dt * 16 + lm, jj * 16, half);
      ctx[dt] = wmma_bf16(P, bvv, ctx[dt]);
    }
    __builtin_amdgcn_wave_barrier();
  }
#pragma unroll
  for (int off = 1; off < 16; off <<= 1)
#pragma unroll
    for (int v = 0; v < 8; ++v) rsum[v] += __shfl_xor(rsum[v], off, 32);
  float rinv[8];
#pragma unroll
  for (int v = 0; v < 8; ++v) rinv[v] = 1.f / rsum[v];

  // context -> [B*S, D_MODEL] bf16
  const int bb = bh >> 4, h = bh & 15;
#pragma unroll
  for (int dt = 0; dt < 4; ++dt)
#pragma unroll
    for (int v = 0; v < 8; ++v) {
      int m = v + half * 8;
      ctxb[((size_t)(bb * SEQ + q0 + m)) * D_MODEL + h * D_K + dt * 16 + lm] =
          f2bf(ctx[dt][v] * rinv[v]);
    }

  // ---- pass C: recompute scores, write normalized weights (paired, two-phase) ----
  e0 = load_b32x16(Kp, D_K, lm, 0, half);
  e1 = load_b32x16(Kp, D_K, lm, 32, half);
  for (int j = 0; j <= qt; j += 2) {
    const int ja = (j + 1 <= qt) ? j + 1 : j;
    bf16x16 o0 = load_b32x16(Kp, D_K, ja * 16 + lm, 0, half);
    bf16x16 o1 = load_b32x16(Kp, D_K, ja * 16 + lm, 32, half);
    f32x8 s;
#pragma unroll
    for (int v = 0; v < 8; ++v) s[v] = 0.f;
    s = wmma_bf16(aq0, e0, s);
    s = wmma_bf16(aq1, e1, s);
#pragma unroll
    for (int v = 0; v < 8; ++v) {
      int m = v + half * 8, kc = j * 16 + lm;
      float e = (kc <= q0 + m) ? __expf(s[v] * inv_scale - rmax[v]) : 0.f;
      AW[(size_t)m * SEQ + kc] = e * rinv[v];
    }
    const int jb = (j + 2 <= qt) ? j + 2 : j;
    e0 = load_b32x16(Kp, D_K, jb * 16 + lm, 0, half);
    e1 = load_b32x16(Kp, D_K, jb * 16 + lm, 32, half);
    if (j + 1 <= qt) {                    // wave-uniform branch
      f32x8 s2;
#pragma unroll
      for (int v = 0; v < 8; ++v) s2[v] = 0.f;
      s2 = wmma_bf16(aq0, o0, s2);
      s2 = wmma_bf16(aq1, o1, s2);
#pragma unroll
      for (int v = 0; v < 8; ++v) {
        int m = v + half * 8, kc = (j + 1) * 16 + lm;
        float e = (kc <= q0 + m) ? __expf(s2[v] * inv_scale - rmax[v]) : 0.f;
        AW[(size_t)m * SEQ + kc] = e * rinv[v];
      }
    }
  }
  // causal upper triangle: b128 zero stores (this is half of the dominant stream)
  {
    const float4 zero = make_float4(0.f, 0.f, 0.f, 0.f);
    const int z0 = (qt + 1) * 16;           // first all-zero column (16-aligned)
#pragma unroll 1
    for (int m = 0; m < 16; ++m) {
      float* rowp = AW + (size_t)m * SEQ;
      for (int c = z0 + lane * 4; c < SEQ; c += 128)
        *(float4*)(rowp + c) = zero;
    }
  }
}

// ---------------- output GEMM (fp32 store, two-phase pipelined) ----------------
__global__ void __launch_bounds__(256)
k_out_gemm(const __bf16* __restrict__ ctxb, const __bf16* __restrict__ WoT,
           const float* __restrict__ bo, float* __restrict__ out) {
  const int lane = threadIdx.x & 31, wave = threadIdx.x >> 5;
  const int half = lane >> 4, lm = lane & 15;
  const int r0 = blockIdx.x * 64 + (wave & 3) * 16;
  const int c0 = blockIdx.y * 128 + (wave >> 2) * 64;
  f32x8 acc[4];
#pragma unroll
  for (int t = 0; t < 4; ++t)
#pragma unroll
    for (int v = 0; v < 8; ++v) acc[t][v] = 0.f;

  bf16x16 a0 = load_a16x32(ctxb, D_MODEL, r0 + lm, 0, half);
  bf16x16 b0[4];
#pragma unroll
  for (int t = 0; t < 4; ++t) b0[t] = load_b32x16(WoT, D_MODEL, c0 + t * 16 + lm, 0, half);

  for (int k0 = 0; k0 < D_MODEL; k0 += 64) {
    const int k1 = k0 + 32;
    bf16x16 a1 = load_a16x32(ctxb, D_MODEL, r0 + lm, k1, half);
    bf16x16 b1[4];
#pragma unroll
    for (int t = 0; t < 4; ++t)
      b1[t] = load_b32x16(WoT, D_MODEL, c0 + t * 16 + lm, k1, half);
#pragma unroll
    for (int t = 0; t < 4; ++t) acc[t] = wmma_bf16(a0, b0[t], acc[t]);
    const int k2 = (k0 + 64 < D_MODEL) ? k0 + 64 : k0;
    a0 = load_a16x32(ctxb, D_MODEL, r0 + lm, k2, half);
#pragma unroll
    for (int t = 0; t < 4; ++t)
      b0[t] = load_b32x16(WoT, D_MODEL, c0 + t * 16 + lm, k2, half);
#pragma unroll
    for (int t = 0; t < 4; ++t) acc[t] = wmma_bf16(a1, b1[t], acc[t]);
  }
#pragma unroll
  for (int t = 0; t < 4; ++t) {
    int oc = c0 + t * 16 + lm;
    float bs = bo[oc];
#pragma unroll
    for (int v = 0; v < 8; ++v) {
      int gr = r0 + v + half * 8;
      out[(size_t)gr * D_MODEL + oc] = acc[t][v] + bs;
    }
  }
}

extern "C" void kernel_launch(void* const* d_in, const int* in_sizes, int n_in,
                              void* d_out, int out_size, void* d_ws, size_t ws_size,
                              hipStream_t stream) {
  (void)in_sizes; (void)n_in; (void)out_size; (void)ws_size;
  const float* x  = (const float*)d_in[0];
  const float* Wq = (const float*)d_in[1];
  const float* bq = (const float*)d_in[2];
  const float* Wk = (const float*)d_in[3];
  const float* bk = (const float*)d_in[4];
  const float* Wv = (const float*)d_in[5];
  const float* bv = (const float*)d_in[6];
  const float* Wo = (const float*)d_in[7];
  const float* bo = (const float*)d_in[8];
  float* out = (float*)d_out;
  float* aw  = out + (size_t)OUT_ELEMS;   // attention weights region of d_out

  // workspace layout (48 MiB total)
  char* ws = (char*)d_ws;
  __bf16* xb    = (__bf16*)(ws);                       //  8 MiB: x bf16
  __bf16* WTall = (__bf16*)(ws + (8u  << 20));         //  6 MiB: WqT,WkT,WvT bf16
  __bf16* WoT   = (__bf16*)(ws + (14u << 20));         //  2 MiB
  __bf16* Qb    = (__bf16*)(ws + (16u << 20));         //  8 MiB [B,H,S,dk]
  __bf16* Kb    = (__bf16*)(ws + (24u << 20));         //  8 MiB [B,H,S,dk]
  __bf16* VTb   = (__bf16*)(ws + (32u << 20));         //  8 MiB [B,H,dk,S]
  __bf16* ctxb  = (__bf16*)(ws + (40u << 20));         //  8 MiB [B*S,D]

  const size_t D2 = (size_t)D_MODEL * D_MODEL;
  k_cvt_bf16<<<OUT_ELEMS / 256, 256, 0, stream>>>(x, xb, OUT_ELEMS);
  k_transpose_bf16<<<(int)(D2 / 256), 256, 0, stream>>>(Wq, WTall);
  k_transpose_bf16<<<(int)(D2 / 256), 256, 0, stream>>>(Wk, WTall + D2);
  k_transpose_bf16<<<(int)(D2 / 256), 256, 0, stream>>>(Wv, WTall + 2 * D2);
  k_transpose_bf16<<<(int)(D2 / 256), 256, 0, stream>>>(Wo, WoT);
  k_qkv_gemm<<<dim3(M_ROWS / 64, D_MODEL / 128, 3), 256, 0, stream>>>(
      xb, WTall, bq, bk, bv, Qb, Kb, VTb);
  k_attn<<<(BATCH * NUM_HEADS * (SEQ / 16)) / 8, 256, 0, stream>>>(Qb, Kb, VTb, aw, ctxb);
  k_out_gemm<<<dim3(M_ROWS / 64, D_MODEL / 128), 256, 0, stream>>>(ctxb, WoT, bo, out);
}
